// FlumenHead_62508954026202
// MI455X (gfx1250) — compile-verified, matrix operand order checked by reading, not run on hardware
//
#include <hip/hip_runtime.h>
#include <hip/hip_bf16.h>

// ---------------------------------------------------------------------------
// Problem constants (from reference): B=64, S=2048, SD=32, F=1024, H=512, D=9
// ---------------------------------------------------------------------------
#define BB 64
#define SS 2048
#define SD 32
#define FF 1024
#define HH 512
#define DD 9
#define G4F 4096          // 4*F gate columns
#define NWG 64            // persistent workgroups, each owns 16 h-columns
#define TPB 256           // 8 wave32 per WG

typedef __attribute__((ext_vector_type(16))) __bf16 v16bf;
typedef __attribute__((ext_vector_type(8)))  float  v8f;

// ---- workspace layout (bytes) ---------------------------------------------
#define WS_BAR   0u                         // 256 B   barrier counter
#define WS_T0    1024u                      // 131072  enc hidden 0 [64,512] f32
#define WS_T1    (WS_T0 + 131072u)          // 131072  enc hidden 1 [64,512] f32
#define WS_H0    (WS_T1 + 131072u)          // 262144  h0 [64,1024] f32
#define WS_WSW   (WS_H0 + 262144u)          // 8 MB    w_hh bf16 swizzled
#define WS_A0    (WS_WSW + 8388608u)        // 131072  h bf16 swizzled buf 0
#define WS_A1    (WS_A0 + 131072u)          // 131072  h bf16 swizzled buf 1

// fragment mapping for a 16(row) x 32(K) bf16 tile (A-layout, §7.12.2):
//   element e (0..15) of lane L (0..31)  ->  row = L&15
//   K = (e&7) + 16*(e>>3) + 8*(L>>4)
// inverse: lane = row + 16*((K>>3)&1) ; e = (K&7) + 8*((K>>4)&1)

__device__ __forceinline__ float sigm_(float x) {
    return 1.0f / (1.0f + __expf(-x));
}
__device__ __forceinline__ float tanh_(float x) {
    // 2*sigmoid(2x)-1 : saturates correctly for |x| large
    return 2.0f / (1.0f + __expf(-2.0f * x)) - 1.0f;
}

// ---------------------------------------------------------------------------
// Encoder MLP layer: out[b,j] = act( sum_k in[b,k]*W[j,k] + bias[j] )
// ---------------------------------------------------------------------------
__global__ void enc_gemm(const float* __restrict__ in, const float* __restrict__ W,
                         const float* __restrict__ bias, float* __restrict__ out,
                         int K, int N, int relu) {
    int idx = blockIdx.x * TPB + threadIdx.x;     // b*N + j
    int b = idx / N, j = idx - b * N;
    const float* ip = in + b * K;
    const float* wp = W + j * K;
    float acc = bias[j];
    for (int k = 0; k < K; ++k) acc = fmaf(ip[k], wp[k], acc);
    if (relu) acc = fmaxf(acc, 0.0f);
    out[idx] = acc;
}

// ---------------------------------------------------------------------------
// Pre-swizzle w_hh (f32 [4096,1024]) -> bf16 WMMA B fragments.
// Tile id = ntile*32 + ktile ; 512 bf16 per tile, lane-major (lane*16 + e).
// ---------------------------------------------------------------------------
__global__ void swizzle_whh(const float* __restrict__ w_hh, __bf16* __restrict__ Wsw) {
    int idx = blockIdx.x * TPB + threadIdx.x;     // < 4096*1024
    int tile = idx >> 9, within = idx & 511;
    int lane = within >> 4, e = within & 15;
    int ntile = tile >> 5, ktile = tile & 31;
    int row = lane & 15;
    int K = (e & 7) + ((e >> 3) << 4) + ((lane >> 4) << 3);
    int j = (ntile << 4) + row;                   // gate column
    int k = (ktile << 5) + K;                     // h column
    Wsw[idx] = (__bf16)w_hh[j * FF + k];
}

// ---------------------------------------------------------------------------
// Seed step 0: h_seq[:,0,:] = h0 ; A0 = swizzled bf16(h0) ; zero barrier.
// ---------------------------------------------------------------------------
__global__ void init_state(const float* __restrict__ h0, float* __restrict__ out,
                           __bf16* __restrict__ A0, unsigned* __restrict__ bar) {
    int idx = blockIdx.x * TPB + threadIdx.x;     // < 64*1024
    int b = idx >> 10, f = idx & 1023;
    float hv = h0[idx];
    out[BB * FF + (size_t)b * SS * FF + f] = hv;  // h_seq[b,0,f]
    int mt = b >> 4, kt = f >> 5, row = b & 15, K = f & 31;
    int lane = row + (((K >> 3) & 1) << 4);
    int e = (K & 7) + (((K >> 4) & 1) << 3);
    A0[((mt * 32 + kt) << 9) + lane * 16 + e] = (__bf16)hv;
    if (idx == 0) *bar = 0u;
}

// ---------------------------------------------------------------------------
// Persistent LSTM scan. 64 WGs x 256 threads. WG wg owns h columns
// [wg*16, wg*16+16). Per step: WMMA GEMM of [64,1024]x[1024, 64 gate cols]
// in bf16 with f32 accumulation, gate fusion, c in registers.
// ---------------------------------------------------------------------------
__global__ void __launch_bounds__(TPB)
lstm_scan(const float* __restrict__ rnn,      // [64,2048,9]
          const float* __restrict__ w_ih,     // [4096,9]
          const float* __restrict__ bias,     // [4096]
          const __bf16* __restrict__ Wsw,     // swizzled weights
          __bf16* __restrict__ A0, __bf16* __restrict__ A1,
          float* __restrict__ out,            // h_last | h_seq
          unsigned* __restrict__ bar) {
    __shared__ float lin[BB][64];     // gate pre-activations for this WG slice
    __shared__ float xs[BB * DD];     // x_t
    __shared__ float wih_s[64 * DD];  // w_ih slice for our 64 gate columns
    __shared__ float bias_s[64];

    const int t    = threadIdx.x;
    const int wg   = blockIdx.x;          // 0..63
    const int fbase = wg * 16;            // first owned h column
    const int wave = t >> 5;              // 0..7
    const int lane = t & 31;
    const int m     = wave & 3;           // M-tile (batch rows m*16..m*16+15)
    const int gpair = wave >> 2;          // 0 -> gates {i,f}, 1 -> gates {g,o}

    // cache w_ih slice + bias for our gate columns (jg = g*16 + fl)
    for (int idx = t; idx < 64 * DD; idx += TPB) {
        int jg = idx / DD, k = idx - jg * DD;
        int j = (jg >> 4) * FF + fbase + (jg & 15);
        wih_s[idx] = w_ih[j * DD + k];
    }
    if (t < 64) {
        int j = (t >> 4) * FF + fbase + (t & 15);
        bias_s[t] = bias[j];
    }

    float creg[4] = {0.f, 0.f, 0.f, 0.f};   // cell state, 4 (b,f) cells/thread
    __bf16* cur = A0;
    __bf16* nxt = A1;

    __syncthreads();

    for (int s = 0; s < SS; ++s) {
        // ---- stage x_t into LDS --------------------------------------------
        for (int idx = t; idx < BB * DD; idx += TPB) {
            int b = idx / DD, k = idx - b * DD;
            xs[idx] = rnn[((size_t)b * SS + s) * DD + k];
        }
        __syncthreads();

        // ---- lin base = bias + x @ w_ih^T ----------------------------------
        #pragma unroll 4
        for (int i = 0; i < 16; ++i) {
            int L = t + TPB * i;          // 0..4095
            int b = L >> 6, jg = L & 63;
            float acc = bias_s[jg];
            #pragma unroll
            for (int k = 0; k < DD; ++k)
                acc = fmaf(xs[b * DD + k], wih_s[jg * DD + k], acc);
            lin[b][jg] = acc;
        }

        // ---- WMMA GEMM: 2 gate tiles per wave, K = 1024 --------------------
        const int n0 = (gpair * 2 + 0) * 64 + wg;   // N-tile of 1st gate
        const int n1 = (gpair * 2 + 1) * 64 + wg;   // N-tile of 2nd gate
        const v16bf* Ap  = reinterpret_cast<const v16bf*>(cur);
        const v16bf* Bp0 = reinterpret_cast<const v16bf*>(Wsw) + (size_t)n0 * 32 * 32;
        const v16bf* Bp1 = reinterpret_cast<const v16bf*>(Wsw) + (size_t)n1 * 32 * 32;
        v8f acc0 = {}, acc1 = {};
        #pragma unroll 4
        for (int kt = 0; kt < 32; ++kt) {
            v16bf a  = Ap[(m * 32 + kt) * 32 + lane];
            v16bf b0 = Bp0[kt * 32 + lane];
            v16bf b1 = Bp1[kt * 32 + lane];
            __builtin_prefetch(&Bp0[(kt + 1) * 32 + lane], 0, 0);  // global_prefetch_b8
            acc0 = __builtin_amdgcn_wmma_f32_16x16x32_bf16(
                false, a, false, b0, (short)0, acc0, false, false);
            acc1 = __builtin_amdgcn_wmma_f32_16x16x32_bf16(
                false, a, false, b1, (short)0, acc1, false, false);
        }
        __syncthreads();   // lin base written by all threads

        // ---- scatter C fragments into LDS (C layout: M = v+8*(lane>=16),
        //      N = lane&15), disjoint tiles per wave -------------------------
        #pragma unroll
        for (int v = 0; v < 8; ++v) {
            int M = v + ((lane >> 4) << 3);
            int N = lane & 15;
            lin[m * 16 + M][gpair * 32 + N]      += acc0[v];
            lin[m * 16 + M][gpair * 32 + 16 + N] += acc1[v];
        }
        __syncthreads();

        // ---- gate fusion + state update ------------------------------------
        #pragma unroll
        for (int i = 0; i < 4; ++i) {
            int cidx = t + TPB * i;       // 0..1023 over 64x16 slice
            int b = cidx >> 4, fl = cidx & 15;
            float ig = lin[b][fl];
            float fg = lin[b][16 + fl];
            float gg = lin[b][32 + fl];
            float og = lin[b][48 + fl];
            float cv = sigm_(fg) * creg[i] + sigm_(ig) * tanh_(gg);
            creg[i] = cv;
            float hv = sigm_(og) * tanh_(cv);
            int f = fbase + fl;
            if (s + 1 < SS)
                out[BB * FF + ((size_t)b * SS + (s + 1)) * FF + f] = hv;  // h_seq
            else
                out[(size_t)b * FF + f] = hv;                            // h_last
            // swizzled bf16 write for next step's A fragments
            int mt = b >> 4, kt2 = f >> 5, row = b & 15, K = f & 31;
            int lane2 = row + (((K >> 3) & 1) << 4);
            int e = (K & 7) + (((K >> 4) & 1) << 3);
            nxt[((mt * 32 + kt2) << 9) + lane2 * 16 + e] = (__bf16)hv;
        }

        // ---- device-wide barrier (monotonic counter) ------------------------
        __threadfence();
        __syncthreads();
        if (t == 0) {
            __hip_atomic_fetch_add(bar, 1u, __ATOMIC_RELEASE, __HIP_MEMORY_SCOPE_AGENT);
            unsigned target = (unsigned)NWG * (unsigned)(s + 1);
            while (__hip_atomic_load(bar, __ATOMIC_ACQUIRE, __HIP_MEMORY_SCOPE_AGENT) < target)
                __builtin_amdgcn_s_sleep(1);
        }
        __syncthreads();

        __bf16* tmp = cur; cur = nxt; nxt = tmp;
    }
}

// ---------------------------------------------------------------------------
extern "C" void kernel_launch(void* const* d_in, const int* in_sizes, int n_in,
                              void* d_out, int out_size, void* d_ws, size_t ws_size,
                              hipStream_t stream) {
    const float* initial_state = (const float*)d_in[0];
    const float* rnn_input     = (const float*)d_in[1];
    const float* enc_w0 = (const float*)d_in[2];
    const float* enc_b0 = (const float*)d_in[3];
    const float* enc_w1 = (const float*)d_in[4];
    const float* enc_b1 = (const float*)d_in[5];
    const float* enc_w2 = (const float*)d_in[6];
    const float* enc_b2 = (const float*)d_in[7];
    const float* w_ih   = (const float*)d_in[8];
    const float* w_hh   = (const float*)d_in[9];
    const float* bias   = (const float*)d_in[10];
    float* out = (float*)d_out;

    char* ws = (char*)d_ws;
    float*    t0  = (float*)(ws + WS_T0);
    float*    t1  = (float*)(ws + WS_T1);
    float*    h0  = (float*)(ws + WS_H0);
    __bf16*   Wsw = (__bf16*)(ws + WS_WSW);
    __bf16*   A0  = (__bf16*)(ws + WS_A0);
    __bf16*   A1  = (__bf16*)(ws + WS_A1);
    unsigned* bar = (unsigned*)(ws + WS_BAR);

    // encoder MLP: 32 -> 512 -> 512 -> 1024
    enc_gemm<<<(BB * HH) / TPB, TPB, 0, stream>>>(initial_state, enc_w0, enc_b0, t0, SD, HH, 1);
    enc_gemm<<<(BB * HH) / TPB, TPB, 0, stream>>>(t0, enc_w1, enc_b1, t1, HH, HH, 1);
    enc_gemm<<<(BB * FF) / TPB, TPB, 0, stream>>>(t1, enc_w2, enc_b2, h0, HH, FF, 0);

    // weight swizzle to bf16 WMMA fragments (L2-resident, reused 2048x)
    swizzle_whh<<<(G4F * FF) / TPB, TPB, 0, stream>>>(w_hh, Wsw);

    // seed h_seq[:,0,:], A0, barrier
    init_state<<<(BB * FF) / TPB, TPB, 0, stream>>>(h0, out, A0, bar);

    // persistent recurrent scan
    lstm_scan<<<NWG, TPB, 0, stream>>>(rnn_input, w_ih, bias, Wsw, A0, A1, out, bar);
}
